// LSTM_54546084659312
// MI455X (gfx1250) — matrix operand order repaired and checked by reading, not compiled, not run on hardware
//
#include <hip/hip_runtime.h>
#include <stdint.h>
#include <math.h>

// ---------------- problem constants ----------------
constexpr int kB = 64, kS = 512, kIN = 256, kH = 512, kOUT = 128;
constexpr int kG4 = 4 * kH;              // 2048 gate rows total
constexpr int NWG = 64, NTHR = 256;      // 64 persistent WGs x 8 waves (wave32)
constexpr int HPW  = kH / NWG;           // 8 h-columns per WG
constexpr int NLOC = 4 * HPW;            // 32 gate rows per WG

typedef __attribute__((ext_vector_type(16))) __bf16      v16bf;
typedef __attribute__((ext_vector_type(8)))  float       v8f;
typedef __attribute__((ext_vector_type(4)))  unsigned    u32x4;
typedef unsigned short ushort_t;

union Frag { u32x4 q[2]; v16bf v; };

// ---------------- workspace layout (ushort elems after 256B barrier header) ---
constexpr size_t N_X    = (size_t)kB * kS * kIN;     // x in bf16
constexpr size_t N_H1   = (size_t)kS * kB * kH;      // layer0 output sequence
constexpr size_t N_WIH0 = (size_t)kG4 * kIN;
constexpr size_t N_WHH  = (size_t)kG4 * kH;
constexpr size_t N_WY   = (size_t)kOUT * kH;
constexpr size_t OFF_X    = 0;
constexpr size_t OFF_H1   = OFF_X + N_X;
constexpr size_t OFF_WIH0 = OFF_H1 + N_H1;
constexpr size_t OFF_WHH0 = OFF_WIH0 + N_WIH0;
constexpr size_t OFF_WIH1 = OFF_WHH0 + N_WHH;
constexpr size_t OFF_WHH1 = OFF_WIH1 + N_WHH;
constexpr size_t OFF_WY   = OFF_WHH1 + N_WHH;
constexpr size_t OFF_HBUF = OFF_WY + N_WY;           // double-buffered h [2][B][H]

constexpr unsigned SMEM_W_BYTES = (unsigned)(NLOC * (kH + kH) * sizeof(ushort_t)); // 64KB
constexpr unsigned SMEM_BYTES   = SMEM_W_BYTES + (unsigned)(kB * NLOC * sizeof(float)); // +8KB gates

// ---------------- small device helpers ----------------
__device__ __forceinline__ ushort_t f2bf(float f) {       // RNE f32 -> bf16
  unsigned u = __float_as_uint(f);
  unsigned r = u + 0x7fffu + ((u >> 16) & 1u);
  return (ushort_t)(r >> 16);
}
__device__ __forceinline__ float sigmf(float x) { return 1.0f / (1.0f + __expf(-x)); }

__device__ __forceinline__ void grid_barrier(unsigned* bar) {
  __syncthreads();
  if (threadIdx.x == 0) {
    __threadfence();
    unsigned prev = __hip_atomic_fetch_add(bar, 1u, __ATOMIC_ACQ_REL, __HIP_MEMORY_SCOPE_AGENT);
    unsigned target = (prev / (unsigned)NWG + 1u) * (unsigned)NWG;
    while (__hip_atomic_load(bar, __ATOMIC_ACQUIRE, __HIP_MEMORY_SCOPE_AGENT) < target)
      __builtin_amdgcn_s_sleep(1);
  }
  __syncthreads();
}

// A fragment: 16x32 bf16, rows rowbase..rowbase+15, K chunk k0 (row-major src)
__device__ __forceinline__ v16bf load_frag_A(const ushort_t* __restrict__ base,
                                             int lda, int rowbase, int k0) {
  int lane = threadIdx.x & 31;
  int row  = rowbase + (lane & 15);
  int koff = (lane < 16) ? 0 : 8;
  const ushort_t* p = base + (size_t)row * lda + (size_t)(k0 + koff);
  Frag f;
  f.q[0] = *(const u32x4*)p;          // K = k0+koff   .. +7
  f.q[1] = *(const u32x4*)(p + 16);   // K = k0+koff+16.. +23
  return f.v;
}
// B fragment: 32x16 bf16 = rows n of W (row-major), K chunk k0
__device__ __forceinline__ v16bf load_frag_B(const ushort_t* __restrict__ base,
                                             int ldb, int nbase, int k0) {
  int lane = threadIdx.x & 31;
  int n    = nbase + (lane & 15);
  int koff = (lane < 16) ? 0 : 8;
  const ushort_t* p = base + (size_t)n * ldb + (size_t)(k0 + koff);
  Frag f;
  f.q[0] = *(const u32x4*)p;          // K = k0+koff   .. +7
  f.q[1] = *(const u32x4*)(p + 16);   // K = k0+koff+16.. +23
  return f.v;
}

// ---------------- one LSTM layer scan ----------------
template <int KIN, bool STORE_H1>
__device__ __forceinline__ void run_layer(
    const ushort_t* __restrict__ aseq, size_t aStepStride, int lda,
    const ushort_t* __restrict__ Wih, const ushort_t* __restrict__ Whh,
    const float* __restrict__ bih, const float* __restrict__ bhh,
    ushort_t* __restrict__ h1out, ushort_t* __restrict__ hbuf,
    ushort_t* smemW, float* smemG, unsigned* bar)
{
  const int tid = threadIdx.x;
  const int wg  = blockIdx.x;
  const int hc0 = wg * HPW;
  constexpr int KTOT = KIN + kH;

  // stage this WG's [Wih | Whh] gate-row slice into LDS (reused for all 512 steps)
  for (int idx = tid; idx < NLOC * KTOT; idx += NTHR) {
    int r = idx / KTOT, k = idx - r * KTOT;
    int grow = (r >> 3) * kH + hc0 + (r & 7);       // gate g = r>>3, col j = r&7
    ushort_t v = (k < KIN) ? Wih[(size_t)grow * KIN + k]
                           : Whh[(size_t)grow * kH + (k - KIN)];
    smemW[(size_t)r * KTOT + k] = v;
  }

  // per-thread cell state: 2 (batch, hcol) pairs; fp32 state in registers
  float cst[2]; float bsum[2][4]; int bidx[2], jidx[2];
#pragma unroll
  for (int q = 0; q < 2; ++q) {
    int p = tid * 2 + q;
    bidx[q] = p >> 3; jidx[q] = p & 7; cst[q] = 0.f;
#pragma unroll
    for (int g = 0; g < 4; ++g)
      bsum[q][g] = bih[g * kH + hc0 + jidx[q]] + bhh[g * kH + hc0 + jidx[q]];
    hbuf[bidx[q] * kH + hc0 + jidx[q]] = 0;                 // zero h(0) both buffers
    hbuf[kB * kH + bidx[q] * kH + hc0 + jidx[q]] = 0;
  }
  grid_barrier(bar);

  const int wv = tid >> 5, lane = tid & 31;
  const int mt = wv & 3;          // 4 M-tiles (64 batches)
  const int nt = wv >> 2;         // 2 N-tiles (32 gate rows)

  for (int t = 0; t < kS; ++t) {
    const ushort_t* Acur  = aseq + (size_t)t * aStepStride;
    const ushort_t* Hcur  = hbuf + (size_t)(t & 1) * kB * kH;
    ushort_t*       Hnext = hbuf + (size_t)((t + 1) & 1) * kB * kH;

    v8f acc = {0.f, 0.f, 0.f, 0.f, 0.f, 0.f, 0.f, 0.f};
#pragma unroll 4
    for (int k0 = 0; k0 < KIN; k0 += 32) {                  // input-gate GEMM
      v16bf a = load_frag_A(Acur, lda, mt * 16, k0);
      v16bf b = load_frag_B(smemW, KTOT, nt * 16, k0);
      acc = __builtin_amdgcn_wmma_f32_16x16x32_bf16(false, a, false, b,
                                                    (short)0, acc, false, false);
    }
#pragma unroll 4
    for (int k0 = 0; k0 < kH; k0 += 32) {                   // recurrent GEMM
      v16bf a = load_frag_A(Hcur, kH, mt * 16, k0);
      v16bf b = load_frag_B(smemW, KTOT, nt * 16, KIN + k0);
      acc = __builtin_amdgcn_wmma_f32_16x16x32_bf16(false, a, false, b,
                                                    (short)0, acc, false, false);
    }
    {   // spill C tile to LDS gate buffer (D layout: M = r + (lane<16?0:8))
      int n = nt * 16 + (lane & 15);
      int mbase = mt * 16 + ((lane < 16) ? 0 : 8);
#pragma unroll
      for (int r = 0; r < 8; ++r) smemG[(mbase + r) * NLOC + n] = acc[r];
    }
    __syncthreads();
#pragma unroll
    for (int q = 0; q < 2; ++q) {                           // LSTM cell update
      int b = bidx[q], j = jidx[q];
      const float* gr = smemG + b * NLOC;
      float iv = sigmf(gr[j]      + bsum[q][0]);
      float fv = sigmf(gr[8 + j]  + bsum[q][1]);
      float gv = tanhf(gr[16 + j] + bsum[q][2]);
      float ov = sigmf(gr[24 + j] + bsum[q][3]);
      float c  = fv * cst[q] + iv * gv;
      cst[q] = c;
      ushort_t hb = f2bf(ov * tanhf(c));
      Hnext[b * kH + hc0 + j] = hb;
      if constexpr (STORE_H1)
        h1out[(size_t)t * kB * kH + b * kH + hc0 + j] = hb;
    }
    grid_barrier(bar);   // publish h(t+1); also guards smemG reuse
  }
}

// ---------------- fused persistent kernel ----------------
__global__ void __launch_bounds__(NTHR, 1)
lstm_fused(const float* __restrict__ x,
           const float* __restrict__ Wih0, const float* __restrict__ Whh0,
           const float* __restrict__ bih0, const float* __restrict__ bhh0,
           const float* __restrict__ Wih1, const float* __restrict__ Whh1,
           const float* __restrict__ bih1, const float* __restrict__ bhh1,
           const float* __restrict__ Wy,   const float* __restrict__ by,
           float* __restrict__ out, unsigned char* __restrict__ wsb)
{
  extern __shared__ char smem[];
  ushort_t* smemW = (ushort_t*)smem;
  float*    smemG = (float*)(smem + SMEM_W_BYTES);
  unsigned* bar   = (unsigned*)wsb;                 // zeroed by hipMemsetAsync
  ushort_t* wse   = (ushort_t*)(wsb + 256);

  ushort_t* xbf  = wse + OFF_X;
  ushort_t* h1   = wse + OFF_H1;
  ushort_t* wih0 = wse + OFF_WIH0;
  ushort_t* whh0 = wse + OFF_WHH0;
  ushort_t* wih1 = wse + OFF_WIH1;
  ushort_t* whh1 = wse + OFF_WHH1;
  ushort_t* wy   = wse + OFF_WY;
  ushort_t* hbuf = wse + OFF_HBUF;

  // phase 0: fp32 -> bf16 staging of activations + weights (grid-stride)
  {
    size_t gid = (size_t)blockIdx.x * NTHR + threadIdx.x;
    const size_t T = (size_t)NWG * NTHR;
    for (size_t i = gid; i < N_X;    i += T) xbf[i]  = f2bf(x[i]);
    for (size_t i = gid; i < N_WIH0; i += T) wih0[i] = f2bf(Wih0[i]);
    for (size_t i = gid; i < N_WHH;  i += T) whh0[i] = f2bf(Whh0[i]);
    for (size_t i = gid; i < N_WHH;  i += T) wih1[i] = f2bf(Wih1[i]);
    for (size_t i = gid; i < N_WHH;  i += T) whh1[i] = f2bf(Whh1[i]);
    for (size_t i = gid; i < N_WY;   i += T) wy[i]   = f2bf(Wy[i]);
  }
  grid_barrier(bar);

  // layer 0: A rows = x[b][t][:]  (row stride S*IN, step base t*IN)
  run_layer<kIN, true >(xbf, (size_t)kIN, kS * kIN,
                        wih0, whh0, bih0, bhh0, h1, hbuf, smemW, smemG, bar);
  // layer 1: A rows = h1[t][b][:]
  run_layer<kH,  false>(h1, (size_t)kB * kH, kH,
                        wih1, whh1, bih1, bhh1, nullptr, hbuf, smemW, smemG, bar);

  // final projection out = h2_last @ Wy^T + by  (64x128, WGs 0..3, 1 tile/wave)
  if (blockIdx.x < 4) {
    const ushort_t* hf = hbuf;                 // S even -> final h in buffer 0
    int wv = threadIdx.x >> 5, lane = threadIdx.x & 31;
    int mt = wv >> 1;
    int nt = blockIdx.x * 2 + (wv & 1);
    v8f acc = {0.f, 0.f, 0.f, 0.f, 0.f, 0.f, 0.f, 0.f};
#pragma unroll 4
    for (int k0 = 0; k0 < kH; k0 += 32) {
      v16bf a = load_frag_A(hf, kH, mt * 16, k0);
      v16bf b = load_frag_B(wy, kH, nt * 16, k0);
      acc = __builtin_amdgcn_wmma_f32_16x16x32_bf16(false, a, false, b,
                                                    (short)0, acc, false, false);
    }
    int n = nt * 16 + (lane & 15);
    float bv = by[n];
    int mbase = mt * 16 + ((lane < 16) ? 0 : 8);
#pragma unroll
    for (int r = 0; r < 8; ++r) out[(mbase + r) * kOUT + n] = acc[r] + bv;
  }
}

extern "C" void kernel_launch(void* const* d_in, const int* in_sizes, int n_in,
                              void* d_out, int out_size, void* d_ws, size_t ws_size,
                              hipStream_t stream) {
  (void)in_sizes; (void)n_in; (void)out_size; (void)ws_size;
  // reset grid-barrier header so graph replays are deterministic
  hipMemsetAsync(d_ws, 0, 256, stream);
  hipFuncSetAttribute(reinterpret_cast<const void*>(lstm_fused),
                      hipFuncAttributeMaxDynamicSharedMemorySize, (int)SMEM_BYTES);
  lstm_fused<<<NWG, NTHR, SMEM_BYTES, stream>>>(
      (const float*)d_in[0],
      (const float*)d_in[1], (const float*)d_in[2],
      (const float*)d_in[3], (const float*)d_in[4],
      (const float*)d_in[5], (const float*)d_in[6],
      (const float*)d_in[7], (const float*)d_in[8],
      (const float*)d_in[9], (const float*)d_in[10],
      (float*)d_out, (unsigned char*)d_ws);
}
